// DimeNetPlusPlus_11321533792783
// MI455X (gfx1250) — matrix operand order repaired and checked by reading, not compiled, not run on hardware
//
#include <hip/hip_runtime.h>
#include <hip/hip_bf16.h>

// Problem constants (fixed by the reference)
#define E_CNT 120000
#define T_CNT 800000
#define N_CNT 12000
#define H     128
#define INTC  64
#define BE    8
#define RR    6
#define SRR   42
#define OE    256

typedef __attribute__((ext_vector_type(16))) __bf16 v16bf;
typedef __attribute__((ext_vector_type(4)))  __bf16 v4bf;
typedef __attribute__((ext_vector_type(8)))  float  v8f;

// Fast silu: x * rcp(1 + e^-x)
__device__ __forceinline__ float silu_f(float x) {
  return x * __builtin_amdgcn_rcpf(1.0f + __expf(-x));
}

// ---------------- async global<->LDS copies (ASYNCcnt tracked) -------------
__device__ __forceinline__ void async_load_b128(unsigned lds_off, unsigned long long gaddr) {
  asm volatile("global_load_async_to_lds_b128 %0, %1, off"
               :: "v"(lds_off), "v"(gaddr) : "memory");
}
__device__ __forceinline__ void async_store_b128(unsigned long long gaddr, unsigned lds_off) {
  asm volatile("global_store_async_from_lds_b128 %0, %1, off"
               :: "v"(gaddr), "v"(lds_off) : "memory");
}
__device__ __forceinline__ void wait_async0() {
#if __has_builtin(__builtin_amdgcn_s_wait_asynccnt)
  __builtin_amdgcn_s_wait_asynccnt(0);
#else
  asm volatile("s_wait_asynccnt 0" ::: "memory");
#endif
}

// ---------------------------------------------------------------------------
// Tile GEMM:  C(128 x ncols) = A(128 x K) * W(K x ncols), bf16 in, f32 acc.
// As: bf16 [128][LDKA] (M x K); Ws: bf16 [ncols][LDKB] (W^T).
// 256-thread block = 8 waves; wave w owns M-rows [16w, 16w+16).
// Fragment layouts per CDNA5 ISA 7.12.2. Epilogues touch LDS only.
// ---------------------------------------------------------------------------
template<int LDKA, int LDKB, typename Epi>
__device__ __forceinline__ void gemm_tile(const __bf16* __restrict__ As,
                                          const __bf16* __restrict__ Ws,
                                          int K, int ncols, Epi&& epi) {
  const int lane = threadIdx.x & 31;
  const int wave = threadIdx.x >> 5;
  const int mrow = wave * 16;
  const int lm   = lane & 15;
  const int hs   = lane >> 4;
  for (int nt = 0; nt < ncols; nt += 16) {
    v8f acc = {};
    for (int kb = 0; kb < K; kb += 32) {
      v16bf a, b;
      const __bf16* ap = As + (mrow + lm) * LDKA + kb + hs * 8;
#pragma unroll
      for (int i = 0; i < 8; ++i) { a[i] = ap[i]; a[i + 8] = ap[16 + i]; }
      const __bf16* bp = Ws + (nt + lm) * LDKB + kb + hs * 8;
#pragma unroll
      for (int i = 0; i < 8; ++i) { b[i] = bp[i]; b[i + 8] = bp[16 + i]; }
      acc = __builtin_amdgcn_wmma_f32_16x16x32_bf16(false, a, false, b,
                                                    (short)0, acc, false, false);
    }
#pragma unroll
    for (int j = 0; j < 8; ++j)
      epi(mrow + j + hs * 8, nt + lm, acc[j]);
  }
}

// ---- cooperative tile movers (256 threads), all 128-bit accesses ----------

template<int LD, int COLS>
__device__ __forceinline__ void load_tile_bf16(__bf16* dst, const float* __restrict__ src,
                                               int e0, int nrows_total) {
  constexpr int C4 = COLS / 4;
  if (e0 + 128 <= nrows_total) {
    const float4* s = (const float4*)(src + (size_t)e0 * COLS);
    for (int idx = threadIdx.x; idx < 128 * C4; idx += 256) {
      int r = idx / C4, c4 = idx - r * C4;
      float4 v = s[idx];
      v4bf o; o[0] = (__bf16)v.x; o[1] = (__bf16)v.y; o[2] = (__bf16)v.z; o[3] = (__bf16)v.w;
      *(v4bf*)(dst + r * LD + c4 * 4) = o;
    }
  } else {
    for (int idx = threadIdx.x; idx < 128 * C4; idx += 256) {
      int r = idx / C4, c4 = idx - r * C4;
      float4 v = make_float4(0.f, 0.f, 0.f, 0.f);
      if (e0 + r < nrows_total)
        v = ((const float4*)(src + (size_t)(e0 + r) * COLS))[c4];
      v4bf o; o[0] = (__bf16)v.x; o[1] = (__bf16)v.y; o[2] = (__bf16)v.z; o[3] = (__bf16)v.w;
      *(v4bf*)(dst + r * LD + c4 * 4) = o;
    }
  }
}

// global f32 tile -> LDS f32 (sync fallback path)
template<int COLS>
__device__ __forceinline__ void load_tile_f32(float* dst, const float* __restrict__ src,
                                              int e0, int nrows_total) {
  constexpr int C4 = COLS / 4;
  for (int idx = threadIdx.x; idx < 128 * C4; idx += 256) {
    int r = idx / C4, c4 = idx - r * C4;
    float4 v = make_float4(0.f, 0.f, 0.f, 0.f);
    if (e0 + r < nrows_total)
      v = ((const float4*)(src + (size_t)(e0 + r) * COLS))[c4];
    ((float4*)dst)[idx] = v;
  }
}

// global f32 tile -> LDS f32 via async DMA (full-tile fast path)
template<int COLS>
__device__ __forceinline__ void async_load_tile_f32(float* dst, const float* __restrict__ src,
                                                    int e0, int nrows_total) {
  constexpr int C4 = COLS / 4;
  if (e0 + 128 <= nrows_total) {
    const float4* s = (const float4*)(src + (size_t)e0 * COLS);
    for (int idx = threadIdx.x; idx < 128 * C4; idx += 256)
      async_load_b128((unsigned)(uintptr_t)(dst + idx * 4),
                      (unsigned long long)(uintptr_t)(s + idx));
  } else {
    load_tile_f32<COLS>(dst, src, e0, nrows_total);
  }
}

// LDS f32 -> global f32 (sync guarded path)
template<int COLS>
__device__ __forceinline__ void store_tile_f32(float* __restrict__ dst, const float* src,
                                               int e0, int nrows_total) {
  constexpr int C4 = COLS / 4;
  for (int idx = threadIdx.x; idx < 128 * C4; idx += 256) {
    int r = idx / C4, c4 = idx - r * C4;
    if (e0 + r < nrows_total)
      ((float4*)(dst + (size_t)(e0 + r) * COLS))[c4] = ((const float4*)src)[idx];
  }
}

// LDS f32 -> global f32 via async DMA (full-tile fast path).
// NOTE: caller must wait_async0() before overwriting src (thread->element
// mapping differs from the reuse loops, so a __syncthreads() must follow).
template<int COLS>
__device__ __forceinline__ void async_store_tile_f32(float* __restrict__ dst, const float* src,
                                                     int e0, int nrows_total) {
  constexpr int C4 = COLS / 4;
  if (e0 + 128 <= nrows_total) {
    float4* d = (float4*)(dst + (size_t)e0 * COLS);
    for (int idx = threadIdx.x; idx < 128 * C4; idx += 256)
      async_store_b128((unsigned long long)(uintptr_t)(d + idx),
                       (unsigned)(uintptr_t)(src + idx * 4));
  } else {
    store_tile_f32<COLS>(dst, src, e0, nrows_total);
  }
}

// LDS f32 [128][128] -> LDS bf16 [128][LD]
template<int LD>
__device__ __forceinline__ void lds_f32_to_bf16(__bf16* dst, const float* src) {
  for (int idx = threadIdx.x; idx < 128 * (H / 4); idx += 256) {
    int r = idx >> 5, c4 = idx & 31;
    float4 v = ((const float4*)src)[idx];
    v4bf o; o[0] = (__bf16)v.x; o[1] = (__bf16)v.y; o[2] = (__bf16)v.z; o[3] = (__bf16)v.w;
    *(v4bf*)(dst + r * LD + c4 * 4) = o;
  }
}

// W[K][ncols] row-major -> Ws[n][k] bf16 (transposed)
template<int LD>
__device__ __forceinline__ void load_w_T(__bf16* dst, const float* __restrict__ W,
                                         int K, int ncols) {
  const int nc4 = ncols / 4;
  for (int idx = threadIdx.x; idx < K * nc4; idx += 256) {
    int k = idx / nc4, n4 = (idx - k * nc4) * 4;
    float4 v = ((const float4*)W)[idx];
    dst[(n4 + 0) * LD + k] = (__bf16)v.x;
    dst[(n4 + 1) * LD + k] = (__bf16)v.y;
    dst[(n4 + 2) * LD + k] = (__bf16)v.z;
    dst[(n4 + 3) * LD + k] = (__bf16)v.w;
  }
}

// ---------------------------------------------------------------------------
// Bessel RBF with p=6 envelope
// ---------------------------------------------------------------------------
__global__ void k_rbf(const float* __restrict__ dist, const float* __restrict__ freq,
                      float* __restrict__ rbf) {
  int e = blockIdx.x * blockDim.x + threadIdx.x;
  if (e >= E_CNT) return;
  float d  = dist[e] * 0.2f;
  float d2 = d * d;
  float d5 = d2 * d2 * d;
  float env = __builtin_amdgcn_rcpf(d) - 28.0f * d5 + 48.0f * d5 * d - 21.0f * d5 * d2;
#pragma unroll
  for (int r = 0; r < RR; ++r)
    rbf[(size_t)e * RR + r] = env * __sinf(freq[r] * d);
}

// ---------------------------------------------------------------------------
// Interaction front half (fused, one 128-edge tile per block)
// ---------------------------------------------------------------------------
__global__ void __launch_bounds__(256)
k_edge_front(const float* __restrict__ xe, const float* __restrict__ rbf,
             const float* __restrict__ W_ji, const float* __restrict__ b_ji,
             const float* __restrict__ W_kj, const float* __restrict__ b_kj,
             const float* __restrict__ Wr1, const float* __restrict__ Wr2,
             const float* __restrict__ W_down,
             float* __restrict__ x_ji_out, float* __restrict__ x_down_out) {
  extern __shared__ char smem[];
  __bf16* As  = (__bf16*)smem;              // 128*136 bf16
  __bf16* A2  = As + 128 * 136;             // 128*136 bf16
  __bf16* Ws  = A2 + 128 * 136;             // 128*136 bf16
  float*  Cf  = (float*)(Ws + 128 * 136);   // 128*128 f32 staging
  float*  t8  = Cf + 128 * H;               // 128*8
  float*  wr2 = t8 + 128 * BE;              // 8*128
  float*  bj  = wr2 + BE * H;               // 128
  float*  bk  = bj + H;                     // 128
  const int e0 = blockIdx.x * 128;

  __builtin_prefetch(W_kj, 0, 0);
  __builtin_prefetch(W_down, 0, 0);

  load_tile_bf16<136, H>(As, xe, e0, E_CNT);
  for (int i = threadIdx.x; i < BE * H; i += 256) wr2[i] = Wr2[i];
  if (threadIdx.x < H) { bj[threadIdx.x] = b_ji[threadIdx.x]; bk[threadIdx.x] = b_kj[threadIdx.x]; }
  for (int i = threadIdx.x; i < 128 * BE; i += 256) {   // t8 = rbf @ Wr1 (6x8)
    int r = i >> 3, j = i & 7;
    float s = 0.f;
    if (e0 + r < E_CNT) {
#pragma unroll
      for (int q = 0; q < RR; ++q) s += rbf[(size_t)(e0 + r) * RR + q] * Wr1[q * BE + j];
    }
    t8[i] = s;
  }
  load_w_T<136>(Ws, W_ji, H, H);
  __syncthreads();

  // x_ji = silu(xe @ W_ji + b_ji)
  gemm_tile<136, 136>(As, Ws, H, H, [&](int m, int n, float v) {
    Cf[m * H + n] = silu_f(v + bj[n]);
  });
  __syncthreads();
  async_store_tile_f32<H>(x_ji_out, Cf, e0, E_CNT);   // DMA out, overlaps next stage
  load_w_T<136>(Ws, W_kj, H, H);
  wait_async0();                                      // Cf recycled below
  __syncthreads();
  // rb[m][n] = dot(t8[m][:], wr2[:][n]) staged in Cf
  for (int i = threadIdx.x; i < 128 * H; i += 256) {
    int m = i >> 7, n = i & 127;
    float rb = 0.f;
#pragma unroll
    for (int j = 0; j < BE; ++j) rb += t8[m * BE + j] * wr2[j * H + n];
    Cf[i] = rb;
  }
  __syncthreads();

  // A2 = bf16( silu(xe@W_kj+b) * rb )
  gemm_tile<136, 136>(As, Ws, H, H, [&](int m, int n, float v) {
    A2[m * 136 + n] = (__bf16)(silu_f(v + bk[n]) * Cf[m * H + n]);
  });
  __syncthreads();
  load_w_T<136>(Ws, W_down, H, INTC);
  __syncthreads();

  gemm_tile<136, 136>(A2, Ws, H, INTC, [&](int m, int n, float v) {
    Cf[m * INTC + n] = silu_f(v);
  });
  __syncthreads();
  async_store_tile_f32<INTC>(x_down_out, Cf, e0, E_CNT);  // S_ENDPGM waits idle
}

// ---------------------------------------------------------------------------
// Triplet phase (one wave per triplet)
// ---------------------------------------------------------------------------
__global__ void __launch_bounds__(256)
k_triplet(const float* __restrict__ sbf, const float* __restrict__ Ws1,
          const float* __restrict__ Ws2, const float* __restrict__ x_down,
          const int* __restrict__ idx_kj, const int* __restrict__ idx_ji,
          float* __restrict__ agg) {
  __shared__ float w1[SRR * BE];
  __shared__ float w2[BE * INTC];
  for (int i = threadIdx.x; i < SRR * BE; i += 256)  w1[i] = Ws1[i];
  for (int i = threadIdx.x; i < BE * INTC; i += 256) w2[i] = Ws2[i];
  __syncthreads();
  const int lane = threadIdx.x & 31;
  const int t = (blockIdx.x * blockDim.x + threadIdx.x) >> 5;
  if (t >= T_CNT) return;
  const float* srow = sbf + (size_t)t * SRR;
  float s0 = srow[lane];
  float s1 = (lane < SRR - 32) ? srow[32 + lane] : 0.0f;
  float t8[BE];
#pragma unroll
  for (int j = 0; j < BE; ++j) {
    float p = s0 * w1[lane * BE + j];
    if (lane < SRR - 32) p += s1 * w1[(lane + 32) * BE + j];
#pragma unroll
    for (int off = 16; off > 0; off >>= 1) p += __shfl_xor(p, off, 32);
    t8[j] = p;
  }
  const int kj = idx_kj[t], ji = idx_ji[t];
#pragma unroll
  for (int hh = 0; hh < 2; ++hh) {
    int n = lane + 32 * hh;
    float v = 0.f;
#pragma unroll
    for (int j = 0; j < BE; ++j) v += t8[j] * w2[j * INTC + n];
    v *= x_down[(size_t)kj * INTC + n];
    atomicAdd(&agg[(size_t)ji * INTC + n], v);
  }
}

// ---------------------------------------------------------------------------
// Interaction back half (fused 8-GEMM chain; h tile resident in LDS).
// x_ji and xe skip tiles arrive via async DMA overlapped with bf16 staging.
// ---------------------------------------------------------------------------
__global__ void __launch_bounds__(256)
k_edge_back(const float* __restrict__ agg, const float* __restrict__ x_ji,
            const float* __restrict__ xe_in, const float* __restrict__ W_up,
            const float* __restrict__ W_res, const float* __restrict__ b_res,
            const float* __restrict__ W_lin, const float* __restrict__ b_lin,
            float* __restrict__ xe_out) {
  extern __shared__ char smem[];
  float*  Hs   = (float*)smem;               // 128*128 f32
  float*  Xs   = Hs + 128 * H;               // 128*128 f32 (xe skip)
  __bf16* As   = (__bf16*)(Xs + 128 * H);    // 128*136
  __bf16* Bb   = As + 128 * 136;             // 128*136
  __bf16* Ws   = Bb + 128 * 136;             // 128*136
  float*  bres = (float*)(Ws + 128 * 136);   // 6*128
  float*  blin = bres + 6 * H;               // 128
  const int e0 = blockIdx.x * 128;

  __builtin_prefetch(W_res, 0, 0);
  __builtin_prefetch(W_lin, 0, 0);

  async_load_tile_f32<H>(Hs, x_ji, e0, E_CNT);   // DMA, overlaps staging below
  async_load_tile_f32<H>(Xs, xe_in, e0, E_CNT);  // DMA (used at lin stage)
  load_tile_bf16<136, INTC>(As, agg, e0, E_CNT);
  load_w_T<136>(Ws, W_up, INTC, H);
  for (int i = threadIdx.x; i < 6 * H; i += 256) bres[i] = b_res[i];
  if (threadIdx.x < H) blin[threadIdx.x] = b_lin[threadIdx.x];
  wait_async0();
  __syncthreads();

  // h = x_ji + silu(agg @ W_up)
  gemm_tile<136, 136>(As, Ws, INTC, H, [&](int m, int n, float v) {
    Hs[m * H + n] += silu_f(v);
  });
  __syncthreads();

  auto residual = [&](int r) {
    lds_f32_to_bf16<136>(As, Hs);
    load_w_T<136>(Ws, W_res + (size_t)r * 2 * H * H, H, H);
    __syncthreads();
    const float* br = bres + r * 2 * H;
    gemm_tile<136, 136>(As, Ws, H, H, [&](int m, int n, float v) {
      Bb[m * 136 + n] = (__bf16)silu_f(v + br[n]);
    });
    __syncthreads();
    load_w_T<136>(Ws, W_res + (size_t)r * 2 * H * H + H * H, H, H);
    __syncthreads();
    gemm_tile<136, 136>(Bb, Ws, H, H, [&](int m, int n, float v) {
      Hs[m * H + n] += silu_f(v + br[H + n]);
    });
    __syncthreads();
  };

  residual(0);

  // h = silu(h @ W_lin + b_lin) + xe   (xe read from f32 Xs, DMA'd earlier)
  lds_f32_to_bf16<136>(As, Hs);
  load_w_T<136>(Ws, W_lin, H, H);
  __syncthreads();
  gemm_tile<136, 136>(As, Ws, H, H, [&](int m, int n, float v) {
    Hs[m * H + n] = silu_f(v + blin[n]) + Xs[m * H + n];
  });
  __syncthreads();

  residual(1);
  residual(2);

  async_store_tile_f32<H>(xe_out, Hs, e0, E_CNT);  // S_ENDPGM waits idle
}

// ---------------------------------------------------------------------------
// Output-block gather: nodes[edge_i[e]] += (rbf[e] @ Wo_rbf) * xe[e]
// ---------------------------------------------------------------------------
__global__ void __launch_bounds__(256)
k_outg(const float* __restrict__ rbf, const float* __restrict__ Wo_rbf_b,
       const float* __restrict__ xe, const int* __restrict__ edge_i,
       float* __restrict__ nodes) {
  __shared__ float w[RR * H];
  for (int i = threadIdx.x; i < RR * H / 4; i += 256)
    ((float4*)w)[i] = ((const float4*)Wo_rbf_b)[i];
  __syncthreads();
  const int lane = threadIdx.x & 31;
  const int e = (blockIdx.x * blockDim.x + threadIdx.x) >> 5;
  if (e >= E_CNT) return;
  float r6[RR];
#pragma unroll
  for (int q = 0; q < RR; ++q) r6[q] = rbf[(size_t)e * RR + q];
  const int ni = edge_i[e];
  float4 xv = ((const float4*)(xe + (size_t)e * H))[lane];
  float xc[4] = {xv.x, xv.y, xv.z, xv.w};
#pragma unroll
  for (int c = 0; c < 4; ++c) {
    int n = lane * 4 + c;
    float g = 0.f;
#pragma unroll
    for (int q = 0; q < RR; ++q) g += r6[q] * w[q * H + n];
    atomicAdd(&nodes[(size_t)ni * H + n], g * xc[c]);
  }
}

// ---------------------------------------------------------------------------
// Node MLP: nodes[N,128] @ Wo_up -> 3x silu(@Wo_lin) -> @Wo_out -> P
// ---------------------------------------------------------------------------
__global__ void __launch_bounds__(256)
k_node_mlp(const float* __restrict__ nodes, const float* __restrict__ Wup,
           const float* __restrict__ bup, const float* __restrict__ Wlin,
           const float* __restrict__ blin, const float* __restrict__ Wout,
           float* __restrict__ P, int accumulate) {
  extern __shared__ char smem[];
  __bf16* A0 = (__bf16*)smem;            // 128*264
  __bf16* A1 = A0 + 128 * 264;           // 128*264
  __bf16* Ws = A1 + 128 * 264;           // 128*264
  float*  bs = (float*)(Ws + 128 * 264); // 256 bias staging
  const int n0 = blockIdx.x * 128;

  __builtin_prefetch(Wlin, 0, 0);
  load_tile_bf16<264, H>(A0, nodes, n0, N_CNT);

  for (int half = 0; half < 2; ++half) {
    for (int i = threadIdx.x; i < H * 32; i += 256) {
      int k = i >> 5, n4 = (i & 31) * 4;
      float4 v = *(const float4*)(Wup + (size_t)k * OE + half * 128 + n4);
      Ws[(n4 + 0) * 264 + k] = (__bf16)v.x;
      Ws[(n4 + 1) * 264 + k] = (__bf16)v.y;
      Ws[(n4 + 2) * 264 + k] = (__bf16)v.z;
      Ws[(n4 + 3) * 264 + k] = (__bf16)v.w;
    }
    if (threadIdx.x < 128) bs[threadIdx.x] = bup[half * 128 + threadIdx.x];
    __syncthreads();
    gemm_tile<264, 264>(A0, Ws, H, 128, [&](int m, int n, float v) {
      A1[m * 264 + half * 128 + n] = (__bf16)(v + bs[n]);
    });
    __syncthreads();
  }

  __bf16* cur = A1;
  __bf16* nxt = A0;
  for (int l = 0; l < 3; ++l) {
    const float* Wl = Wlin + (size_t)l * OE * OE;
    const float* bl = blin + l * OE;
    for (int half = 0; half < 2; ++half) {
      for (int i = threadIdx.x; i < OE * 32; i += 256) {
        int k = i >> 5, n4 = (i & 31) * 4;
        float4 v = *(const float4*)(Wl + (size_t)k * OE + half * 128 + n4);
        Ws[(n4 + 0) * 264 + k] = (__bf16)v.x;
        Ws[(n4 + 1) * 264 + k] = (__bf16)v.y;
        Ws[(n4 + 2) * 264 + k] = (__bf16)v.z;
        Ws[(n4 + 3) * 264 + k] = (__bf16)v.w;
      }
      if (threadIdx.x < 128) bs[threadIdx.x] = bl[half * 128 + threadIdx.x];
      __syncthreads();
      gemm_tile<264, 264>(cur, Ws, OE, 128, [&](int m, int n, float v) {
        nxt[m * 264 + half * 128 + n] = (__bf16)silu_f(v + bs[n]);
      });
      __syncthreads();
    }
    __bf16* tmp = cur; cur = nxt; nxt = tmp;
  }

  if (threadIdx.x < 128) {
    int r = threadIdx.x;
    if (n0 + r < N_CNT) {
      float v = 0.f;
      for (int k = 0; k < OE; ++k) v += (float)cur[r * 264 + k] * Wout[k];
      if (accumulate) P[n0 + r] += v;
      else            P[n0 + r]  = v;
    }
  }
}

// ---------------------------------------------------------------------------
extern "C" void kernel_launch(void* const* d_in, const int* in_sizes, int n_in,
                              void* d_out, int out_size, void* d_ws, size_t ws_size,
                              hipStream_t stream) {
  const float* x       = (const float*)d_in[0];
  const float* dist    = (const float*)d_in[1];
  const float* freq    = (const float*)d_in[2];
  const float* sbf     = (const float*)d_in[3];
  const int*   idx_kj  = (const int*)d_in[4];
  const int*   idx_ji  = (const int*)d_in[5];
  const int*   edge_i  = (const int*)d_in[6];
  const float* Wi_rbf1 = (const float*)d_in[8];
  const float* Wi_rbf2 = (const float*)d_in[9];
  const float* Wi_sbf1 = (const float*)d_in[10];
  const float* Wi_sbf2 = (const float*)d_in[11];
  const float* Wi_kj   = (const float*)d_in[12];
  const float* bi_kj   = (const float*)d_in[13];
  const float* Wi_ji   = (const float*)d_in[14];
  const float* bi_ji   = (const float*)d_in[15];
  const float* Wi_down = (const float*)d_in[16];
  const float* Wi_up   = (const float*)d_in[17];
  const float* Wi_res  = (const float*)d_in[18];
  const float* bi_res  = (const float*)d_in[19];
  const float* Wi_lin  = (const float*)d_in[20];
  const float* bi_lin  = (const float*)d_in[21];
  const float* Wo_rbf  = (const float*)d_in[22];
  const float* Wo_up   = (const float*)d_in[23];
  const float* bo_up   = (const float*)d_in[24];
  const float* Wo_lin  = (const float*)d_in[25];
  const float* bo_lin  = (const float*)d_in[26];
  const float* Wo_out  = (const float*)d_in[27];
  float* P = (float*)d_out;

  float* ws = (float*)d_ws;
  size_t off = 0;
  float* rbf    = ws + off; off += (size_t)E_CNT * RR;
  float* x_ji   = ws + off; off += (size_t)E_CNT * H;
  float* x_down = ws + off; off += (size_t)E_CNT * INTC;
  float* agg    = ws + off; off += (size_t)E_CNT * INTC;
  float* xeA    = ws + off; off += (size_t)E_CNT * H;
  float* xeB    = ws + off; off += (size_t)E_CNT * H;
  float* nodes  = ws + off; off += (size_t)N_CNT * H;

  const int SM_FRONT = 3 * 128 * 136 * 2 + 128 * H * 4 + 128 * BE * 4
                     + BE * H * 4 + 2 * H * 4;                          // ~180 KB
  const int SM_BACK  = 2 * 128 * H * 4 + 3 * 128 * 136 * 2 + 7 * H * 4; // ~239 KB
  const int SM_NODE  = 3 * 128 * 264 * 2 + 256 * 4;                     // ~204 KB
  hipFuncSetAttribute((const void*)k_edge_front, hipFuncAttributeMaxDynamicSharedMemorySize, SM_FRONT);
  hipFuncSetAttribute((const void*)k_edge_back,  hipFuncAttributeMaxDynamicSharedMemorySize, SM_BACK);
  hipFuncSetAttribute((const void*)k_node_mlp,   hipFuncAttributeMaxDynamicSharedMemorySize, SM_NODE);

  const int EB   = (E_CNT + 127) / 128;
  const int NBK  = (N_CNT + 127) / 128;
  const int TBLK = (T_CNT + 7) / 8;
  const int OBLK = (E_CNT + 7) / 8;

  k_rbf<<<(E_CNT + 255) / 256, 256, 0, stream>>>(dist, freq, rbf);

  hipMemsetAsync(nodes, 0, (size_t)N_CNT * H * sizeof(float), stream);
  k_outg<<<OBLK, 256, 0, stream>>>(rbf, Wo_rbf, x, edge_i, nodes);
  k_node_mlp<<<NBK, 256, SM_NODE, stream>>>(nodes, Wo_up, bo_up, Wo_lin, bo_lin,
                                            Wo_out, P, 0);

  const float* xe_cur = x;
  float* xe_next = xeA;
  for (int b = 0; b < 2; ++b) {
    k_edge_front<<<EB, 256, SM_FRONT, stream>>>(
        xe_cur, rbf,
        Wi_ji + (size_t)b * H * H, bi_ji + b * H,
        Wi_kj + (size_t)b * H * H, bi_kj + b * H,
        Wi_rbf1 + (size_t)b * RR * BE, Wi_rbf2 + (size_t)b * BE * H,
        Wi_down + (size_t)b * H * INTC,
        x_ji, x_down);

    hipMemsetAsync(agg, 0, (size_t)E_CNT * INTC * sizeof(float), stream);
    k_triplet<<<TBLK, 256, 0, stream>>>(
        sbf, Wi_sbf1 + (size_t)b * SRR * BE, Wi_sbf2 + (size_t)b * BE * INTC,
        x_down, idx_kj, idx_ji, agg);

    k_edge_back<<<EB, 256, SM_BACK, stream>>>(
        agg, x_ji, xe_cur,
        Wi_up + (size_t)b * INTC * H,
        Wi_res + (size_t)b * 3 * 2 * H * H, bi_res + (size_t)b * 3 * 2 * H,
        Wi_lin + (size_t)b * H * H, bi_lin + b * H,
        xe_next);

    hipMemsetAsync(nodes, 0, (size_t)N_CNT * H * sizeof(float), stream);
    k_outg<<<OBLK, 256, 0, stream>>>(rbf, Wo_rbf + (size_t)(b + 1) * RR * H,
                                     xe_next, edge_i, nodes);
    k_node_mlp<<<NBK, 256, SM_NODE, stream>>>(
        nodes, Wo_up + (size_t)(b + 1) * H * OE, bo_up + (b + 1) * OE,
        Wo_lin + (size_t)(b + 1) * 3 * OE * OE, bo_lin + (b + 1) * 3 * OE,
        Wo_out + (size_t)(b + 1) * OE, P, 1);

    xe_cur = xe_next;
    xe_next = xeB;
  }
}
